// GCNLayer_6605659701677
// MI455X (gfx1250) — compile-verified, matrix operand order checked by reading, not run on hardware
//
#include <hip/hip_runtime.h>
#include <hip/hip_bf16.h>

// ---------------------------------------------------------------------------
// GCN layer for MI455X (gfx1250): bf16 WMMA dual-GEMM + streamed edge scatter
// ---------------------------------------------------------------------------

#define DD 128

typedef __bf16 bf16;
typedef bf16  v16bf __attribute__((ext_vector_type(16)));
typedef float v8f   __attribute__((ext_vector_type(8)));

// ---- zero scratch ----------------------------------------------------------
__global__ void zero_kernel(float* __restrict__ p, size_t n) {
    size_t i = (size_t)blockIdx.x * blockDim.x + threadIdx.x;
    size_t stride = (size_t)gridDim.x * blockDim.x;
    for (; i < n; i += stride) p[i] = 0.0f;
}

// ---- degree counts ---------------------------------------------------------
__global__ void degree_kernel(const int* __restrict__ senders,
                              const int* __restrict__ receivers,
                              float* __restrict__ deg_s,
                              float* __restrict__ deg_r, int E) {
    int stride = gridDim.x * blockDim.x;
    for (int i = blockIdx.x * blockDim.x + threadIdx.x; i < E; i += stride) {
        atomicAdd(&deg_s[senders[i]], 1.0f);
        atomicAdd(&deg_r[receivers[i]], 1.0f);
    }
}

// ---- fused dual GEMM: h_pre = nodes@W1+b1 ; conv = (nodes@W2+b2)*rsqrt(deg_s)
// One workgroup = 8 waves, each wave owns a 16-row stripe (128 rows / WG).
// W1^T / W2^T staged in LDS as bf16 so B-fragments are contiguous 16B reads.
__global__ void __launch_bounds__(256)
gemm_dual_kernel(const float* __restrict__ nodes,
                 const float* __restrict__ W1w, const float* __restrict__ W1b,
                 const float* __restrict__ W2w, const float* __restrict__ W2b,
                 const float* __restrict__ deg_s,
                 float* __restrict__ hout, float* __restrict__ conv, int Nn) {
    __shared__ bf16 WT1[DD * DD];   // W1^T, bf16, 32 KB
    __shared__ bf16 WT2[DD * DD];   // W2^T, bf16, 32 KB

    const int tid = threadIdx.x;
    for (int idx = tid; idx < DD * DD; idx += 256) {
        int k = idx >> 7, n = idx & 127;
        WT1[n * DD + k] = (bf16)W1w[idx];
        WT2[n * DD + k] = (bf16)W2w[idx];
    }
    __syncthreads();

    const int wave = tid >> 5;
    const int lane = tid & 31;
    const int nl   = lane & 15;          // column within 16-wide tile
    const int mh   = (lane >> 4) << 3;   // 0 or 8: row-half (C/D) and K-base (A/B)
    const int kb   = mh;
    const int row0 = blockIdx.x * 128 + wave * 16;

    // A-panel: 16 rows x 128 K as 4 bf16 fragments (clamped row for OOB waves;
    // stores are guarded, WMMA needs full EXEC so we never mask lanes).
    int rA = row0 + nl;
    if (rA > Nn - 1) rA = Nn - 1;
    const float* arow = nodes + (size_t)rA * DD;

    v16bf a[4];
#pragma unroll
    for (int c = 0; c < 4; ++c) {
        const float4 f0 = *(const float4*)(arow + 32 * c + kb);
        const float4 f1 = *(const float4*)(arow + 32 * c + kb + 4);
        const float4 f2 = *(const float4*)(arow + 32 * c + kb + 16);
        const float4 f3 = *(const float4*)(arow + 32 * c + kb + 20);
        a[c][0]  = (bf16)f0.x; a[c][1]  = (bf16)f0.y; a[c][2]  = (bf16)f0.z; a[c][3]  = (bf16)f0.w;
        a[c][4]  = (bf16)f1.x; a[c][5]  = (bf16)f1.y; a[c][6]  = (bf16)f1.z; a[c][7]  = (bf16)f1.w;
        a[c][8]  = (bf16)f2.x; a[c][9]  = (bf16)f2.y; a[c][10] = (bf16)f2.z; a[c][11] = (bf16)f2.w;
        a[c][12] = (bf16)f3.x; a[c][13] = (bf16)f3.y; a[c][14] = (bf16)f3.z; a[c][15] = (bf16)f3.w;
    }

#pragma unroll
    for (int j = 0; j < 8; ++j) {        // 8 column tiles of 16
        v8f acc1 = {0.f, 0.f, 0.f, 0.f, 0.f, 0.f, 0.f, 0.f};
        v8f acc2 = {0.f, 0.f, 0.f, 0.f, 0.f, 0.f, 0.f, 0.f};
        const bf16* w1row = &WT1[(16 * j + nl) * DD];
        const bf16* w2row = &WT2[(16 * j + nl) * DD];
#pragma unroll
        for (int c = 0; c < 4; ++c) {    // K chunks of 32
            v16bf bf1, bf2;
#pragma unroll
            for (int e = 0; e < 8; ++e) {
                bf1[e]     = w1row[32 * c + kb + e];
                bf1[e + 8] = w1row[32 * c + kb + 16 + e];
                bf2[e]     = w2row[32 * c + kb + e];
                bf2[e + 8] = w2row[32 * c + kb + 16 + e];
            }
            acc1 = __builtin_amdgcn_wmma_f32_16x16x32_bf16(
                false, a[c], false, bf1, (short)0, acc1, false, false);
            acc2 = __builtin_amdgcn_wmma_f32_16x16x32_bf16(
                false, a[c], false, bf2, (short)0, acc2, false, false);
        }
        const int n = 16 * j + nl;
        const float bias1 = W1b[n];
        const float bias2 = W2b[n];
#pragma unroll
        for (int v = 0; v < 8; ++v) {
            int row = row0 + mh + v;
            if (row < Nn) {
                size_t o = (size_t)row * DD + n;
                hout[o] = acc1[v] + bias1;
                float sc = rsqrtf(fmaxf(deg_s[row], 1.0f));
                conv[o] = (acc2[v] + bias2) * sc;
            }
        }
    }
}

// ---- gvec = globals @ W3 + b3 (1x128, tiny) --------------------------------
__global__ void gvec_kernel(const float* __restrict__ globals_,
                            const float* __restrict__ W3w,
                            const float* __restrict__ W3b,
                            float* __restrict__ gvec) {
    int n = threadIdx.x;
    float s = W3b[n];
    for (int k = 0; k < DD; ++k) s += globals_[k] * W3w[k * DD + n];
    gvec[n] = s;
}

// ---- edge gather / scatter-add: one wave per edge, float4 per lane ---------
__global__ void edge_kernel(const float* __restrict__ conv,
                            const int* __restrict__ senders,
                            const int* __restrict__ receivers,
                            float* __restrict__ agg, int E) {
    const int lane = threadIdx.x & 31;
    const int wid  = (blockIdx.x * blockDim.x + threadIdx.x) >> 5;
    const int nw   = (gridDim.x * blockDim.x) >> 5;
    for (int e = wid; e < E; e += nw) {
        if (e + nw < E) {
            __builtin_prefetch(&senders[e + nw], 0, 0);     // global_prefetch_b8
            __builtin_prefetch(&receivers[e + nw], 0, 0);
        }
        const int s = senders[e];
        const int r = receivers[e];
        const float4 v = *(const float4*)(conv + (size_t)s * DD + lane * 4);
        float* dst = agg + (size_t)r * DD + lane * 4;
        atomicAdd(dst + 0, v.x);
        atomicAdd(dst + 1, v.y);
        atomicAdd(dst + 2, v.z);
        atomicAdd(dst + 3, v.w);
    }
}

// ---- fuse: h = relu(h_pre + agg*rsqrt(deg_r) + gvec) + nodes; col-sum -> an
__global__ void __launch_bounds__(256)
final_fuse_kernel(float* __restrict__ h, const float* __restrict__ nodes,
                  const float* __restrict__ agg, const float* __restrict__ deg_r,
                  const float* __restrict__ gvec, float* __restrict__ an, int Nn) {
    __shared__ float colsum[DD];
    const int t = threadIdx.x;
    if (t < DD) colsum[t] = 0.0f;
    __syncthreads();

    const int col = t & 127;
    const int r0  = blockIdx.x * 2 + (t >> 7);
    const float g = gvec[col];
    float local = 0.0f;
    for (int r = r0; r < Nn; r += gridDim.x * 2) {
        size_t o = (size_t)r * DD + col;
        float sc = rsqrtf(fmaxf(deg_r[r], 1.0f));
        float v = h[o] + agg[o] * sc + g;
        v = fmaxf(v, 0.0f) + nodes[o];
        h[o] = v;
        local += v;
    }
    atomicAdd(&colsum[col], local);
    __syncthreads();
    if (t < DD) atomicAdd(&an[t], colsum[t]);
}

// ---- g_new = globals + relu([an, globals] @ Wg + bg) -----------------------
__global__ void gnew_kernel(const float* __restrict__ an,
                            const float* __restrict__ globals_,
                            const float* __restrict__ Wgw,
                            const float* __restrict__ Wgb,
                            float* __restrict__ gout) {
    int n = threadIdx.x;
    float s = Wgb[n];
    for (int k = 0; k < DD; ++k) s += an[k] * Wgw[k * DD + n];
    for (int k = 0; k < DD; ++k) s += globals_[k] * Wgw[(DD + k) * DD + n];
    gout[n] = globals_[n] + fmaxf(s, 0.0f);
}

// ---------------------------------------------------------------------------
extern "C" void kernel_launch(void* const* d_in, const int* in_sizes, int n_in,
                              void* d_out, int out_size, void* d_ws, size_t ws_size,
                              hipStream_t stream) {
    const float* nodes    = (const float*)d_in[0];
    const float* globals_ = (const float*)d_in[1];
    const int*   senders  = (const int*)d_in[2];
    const int*   receivers= (const int*)d_in[3];
    const float* W1w = (const float*)d_in[4];
    const float* W1b = (const float*)d_in[5];
    const float* W2w = (const float*)d_in[6];
    const float* W2b = (const float*)d_in[7];
    const float* W3w = (const float*)d_in[8];
    const float* W3b = (const float*)d_in[9];
    const float* Wgw = (const float*)d_in[10];
    const float* Wgb = (const float*)d_in[11];

    const int Nn = in_sizes[0] / DD;
    const int E  = in_sizes[2];
    const size_t ND = (size_t)Nn * DD;

    // workspace layout (floats)
    float* ws    = (float*)d_ws;
    float* conv  = ws;                 // ND
    float* agg   = conv + ND;          // ND   (zeroed)
    float* deg_s = agg + ND;           // Nn   (zeroed)
    float* deg_r = deg_s + Nn;         // Nn   (zeroed)
    float* an    = deg_r + Nn;         // 128  (zeroed)
    float* gvec  = an + DD;            // 128

    float* hout = (float*)d_out;       // [Nn, 128]
    float* gout = hout + ND;           // [128]

    const size_t nzero = ND + 2 * (size_t)Nn + DD;   // agg..an contiguous
    zero_kernel<<<2048, 256, 0, stream>>>(agg, nzero);
    degree_kernel<<<1024, 256, 0, stream>>>(senders, receivers, deg_s, deg_r, E);
    gemm_dual_kernel<<<(Nn + 127) / 128, 256, 0, stream>>>(
        nodes, W1w, W1b, W2w, W2b, deg_s, hout, conv, Nn);
    gvec_kernel<<<1, DD, 0, stream>>>(globals_, W3w, W3b, gvec);
    edge_kernel<<<2048, 256, 0, stream>>>(conv, senders, receivers, agg, E);
    final_fuse_kernel<<<1024, 256, 0, stream>>>(hout, nodes, agg, deg_r, gvec, an, Nn);
    gnew_kernel<<<1, DD, 0, stream>>>(an, globals_, Wgw, Wgb, gout);
}
